// Zamba2MambaMixer_36713380446314
// MI455X (gfx1250) — matrix operand order repaired and checked by reading, not compiled, not run on hardware
//
#include <hip/hip_runtime.h>
#include <hip/hip_bf16.h>

// ---------------- problem constants (from reference) ----------------
#define Bb     2
#define Ss     2048
#define Hh     2048
#define Ee     4096
#define PROJ_  8384        // E + CONV_DIM + NH
#define CONVD  4224        // E + 2*G*N
#define NHh    64
#define Pp     64
#define Nn     64
#define CSs    256
#define NCc    8
#define MROWS  4096        // B*S

typedef unsigned short ushort_t;
typedef __attribute__((ext_vector_type(16))) unsigned short v16u;
typedef __attribute__((ext_vector_type(8)))  float v8f;
typedef __bf16 bf16t;
typedef __attribute__((ext_vector_type(16))) bf16t v16bf;

// Native fptrunc -> v_cvt_pk_bf16_f32 on gfx1250 (RTNE).
__device__ __forceinline__ ushort_t f2bf(float f) {
  union { bf16t h; ushort_t u; } c;
  c.h = (bf16t)f;
  return c.u;
}

// fast transcendental helpers (v_exp_f32 / v_rcp_f32 rate paths)
__device__ __forceinline__ float fexp(float x) { return __expf(x); }
__device__ __forceinline__ float silu(float x) {
  return x * __builtin_amdgcn_rcpf(1.f + __expf(-x));
}

__device__ __forceinline__ v8f vzero8() {
  v8f v;
#pragma unroll
  for (int i = 0; i < 8; ++i) v[i] = 0.f;
  return v;
}

__device__ __forceinline__ v8f wmma_bf16(v16u a, v16u b, v8f c) {
  union { v16u u; v16bf h; } A, B;
  A.u = a; B.u = b;
  return __builtin_amdgcn_wmma_f32_16x16x32_bf16(false, A.h, false, B.h,
                                                 (short)0, c, false, false);
}

// A fragment (16x32, bf16) from LDS: rows m (lane&15), k pattern per ISA layout.
__device__ __forceinline__ v16u lds_afrag(const ushort_t* p, int ld, int lane) {
  int m  = lane & 15;
  int kh = (lane >> 4) << 3;
  const ushort_t* row = p + m * ld;
  v16u r;
#pragma unroll
  for (int e = 0; e < 16; ++e) {
    int k = (e & 7) + ((e >> 3) << 4) + kh;
    r[e] = row[k];
  }
  return r;
}

// B fragment (32x16, bf16) from LDS laid out [k][n]: lane->n column, e->k (table order).
__device__ __forceinline__ v16u lds_bfrag(const ushort_t* p, int ld, int lane) {
  int n  = lane & 15;
  int kh = (lane >> 4) << 3;
  v16u r;
#pragma unroll
  for (int e = 0; e < 16; ++e) {
    int k = (e & 7) + ((e >> 3) << 4) + kh;
    r[e] = p[k * ld + n];
  }
  return r;
}

// A fragment directly from fp32 global row-major [m][k].
__device__ __forceinline__ v16u ga_afrag(const float* g, size_t ld, int lane) {
  int m  = lane & 15;
  int kh = (lane >> 4) << 3;
  const float* row = g + (size_t)m * ld;
  v16u r;
#pragma unroll
  for (int e = 0; e < 16; ++e) {
    int k = (e & 7) + ((e >> 3) << 4) + kh;
    r[e] = f2bf(row[k]);
  }
  return r;
}

// B fragment from fp32 global [k][n]: lane->n column, e->k (table order), row k scaled by ds[k*dld].
__device__ __forceinline__ v16u gb_scaled(const float* g, size_t ld, const float* ds, size_t dld, int lane) {
  int n  = lane & 15;
  int kh = (lane >> 4) << 3;
  v16u r;
#pragma unroll
  for (int e = 0; e < 16; ++e) {
    int k = (e & 7) + ((e >> 3) << 4) + kh;
    r[e] = f2bf(g[(size_t)k * ld + n] * ds[(size_t)k * dld]);
  }
  return r;
}

// ---------------- 1/8. generic NT GEMM: C[M,N] = A[M,K] * B[N,K]^T ----------
__global__ __launch_bounds__(256) void gemm_nt_bf16(
    const float* __restrict__ A, const float* __restrict__ Bw,
    float* __restrict__ C, int M, int N, int Kd)
{
  __shared__ ushort_t sA[128 * 32];   // [m][k]
  __shared__ ushort_t sB[32 * 128];   // [k][n]
  const int tid  = threadIdx.x;
  const int lane = tid & 31;
  const int wave = tid >> 5;
  const int m0 = blockIdx.y * 128;
  const int n0 = blockIdx.x * 128;
  const int wm = (wave >> 1) * 32;
  const int wn = (wave & 1) * 64;

  v8f acc[2][4];
#pragma unroll
  for (int i = 0; i < 2; ++i)
#pragma unroll
    for (int t = 0; t < 4; ++t) acc[i][t] = vzero8();

  const int rowA = tid >> 1;           // 0..127
  const int kofs = (tid & 1) * 16;
  const bool okB = (n0 + rowA) < N;
  const float* gA = A + (size_t)(m0 + rowA) * Kd + kofs;
  const float* gB = Bw + (size_t)(n0 + rowA) * Kd + kofs;

  for (int k0 = 0; k0 < Kd; k0 += 32) {
    // prefetch next K tile into cache while we stage this one
    if (k0 + 32 < Kd) {
      __builtin_prefetch(gA + k0 + 32, 0, 1);
      if (okB) __builtin_prefetch(gB + k0 + 32, 0, 1);
    }
    {
      const float* g = gA + k0;
      ushort_t* d = &sA[rowA * 32 + kofs];
#pragma unroll
      for (int i = 0; i < 16; ++i) d[i] = f2bf(g[i]);
    }
    {
      const float* g = gB + k0;
#pragma unroll
      for (int i = 0; i < 16; ++i)
        sB[(kofs + i) * 128 + rowA] = okB ? f2bf(g[i]) : (ushort_t)0;
    }
    __syncthreads();

    v16u af0 = lds_afrag(&sA[(wm + 0) * 32], 32, lane);
    v16u af1 = lds_afrag(&sA[(wm + 16) * 32], 32, lane);
#pragma unroll
    for (int t = 0; t < 4; ++t) {
      v16u bf = lds_bfrag(&sB[wn + t * 16], 128, lane);
      acc[0][t] = wmma_bf16(af0, bf, acc[0][t]);
      acc[1][t] = wmma_bf16(af1, bf, acc[1][t]);
    }
    __syncthreads();
  }

#pragma unroll
  for (int i = 0; i < 2; ++i)
#pragma unroll
    for (int t = 0; t < 4; ++t)
#pragma unroll
      for (int r = 0; r < 8; ++r) {
        int m = m0 + wm + 16 * i + r + ((lane >> 4) << 3);
        int n = n0 + wn + 16 * t + (lane & 15);
        if (n < N) C[(size_t)m * N + n] = acc[i][t][r];
      }
}

// ---------------- 2. depthwise causal conv (K=4) + SiLU ---------------------
__global__ void conv_silu_k(const float* __restrict__ proj,
                            const float* __restrict__ cw,
                            const float* __restrict__ cb,
                            float* __restrict__ out)
{
  size_t idx = (size_t)blockIdx.x * blockDim.x + threadIdx.x;
  if (idx >= (size_t)Bb * Ss * CONVD) return;
  int c = (int)(idx % CONVD);
  int s = (int)((idx / CONVD) % Ss);
  int b = (int)(idx / ((size_t)CONVD * Ss));
  float acc = cb[c];
#pragma unroll
  for (int k = 0; k < 4; ++k) {
    int sp = s - 3 + k;
    if (sp >= 0)
      acc += proj[((size_t)(b * Ss + sp)) * PROJ_ + Ee + c] * cw[c * 4 + k];
  }
  out[idx] = silu(acc);
}

// ---------------- 3. softplus(dt), cumsum(A*dt) per chunk -------------------
__global__ void dt_scan_k(const float* __restrict__ proj,
                          const float* __restrict__ dt_bias,
                          const float* __restrict__ A_log,
                          float* __restrict__ dt,
                          float* __restrict__ acs,
                          float* __restrict__ csum)
{
  int t = blockIdx.x * blockDim.x + threadIdx.x;
  if (t >= Bb * NCc * NHh) return;
  int h = t & 63, c = (t >> 6) & 7, b = t >> 9;
  float Aneg = -fexp(A_log[h]);
  float bias = dt_bias[h];
  float run  = 0.f;
  int base = (b * NCc + c) * NHh + h;
  for (int l = 0; l < CSs; ++l) {
    size_t row = (size_t)(b * Ss + c * CSs + l);
    float x = proj[row * PROJ_ + (Ee + CONVD) + h] + bias;
    float d = (x > 20.f) ? x : log1pf(fexp(x));
    dt[row * NHh + h] = d;
    run += Aneg * d;
    acs[(size_t)base * CSs + l] = run;
  }
  csum[base] = run;
}

// ---------------- 4. per-chunk local states (64x64 = hdt^T * Bdec) ----------
__global__ __launch_bounds__(128) void chunk_states_k(
    const float* __restrict__ conv, const float* __restrict__ dt,
    const float* __restrict__ acs, const float* __restrict__ csum,
    float* __restrict__ lst)
{
  extern __shared__ __align__(16) char smem[];
  ushort_t* sHt = (ushort_t*)smem;      // [p=64][l=256]  (h*dt transposed)
  ushort_t* sB  = sHt + 64 * 256;       // [l=256][n=64]  (B * decay)

  int bx = blockIdx.x;
  int h = bx & 63, c = (bx >> 6) & 7, b = bx >> 9;
  int tid = threadIdx.x, lane = tid & 31, wave = tid >> 5;
  int abase = (b * NCc + c) * NHh + h;
  float clast = csum[abase];
  size_t rowbase = (size_t)(b * Ss + c * CSs);

  for (int i = 0; i < 128; ++i) {
    int idx = i * 128 + tid;
    int l = idx >> 6, p = idx & 63;
    size_t row = rowbase + l;
    float dscale = dt[row * NHh + h];
    sHt[p * 256 + l] = f2bf(conv[row * CONVD + h * Pp + p] * dscale);
    float dec = fexp(clast - acs[(size_t)abase * CSs + l]);
    sB[l * 64 + p] = f2bf(conv[row * CONVD + Ee + p] * dec);  // p == n here
  }
  __syncthreads();

  v8f acc[4];
#pragma unroll
  for (int t = 0; t < 4; ++t) acc[t] = vzero8();

  for (int k0 = 0; k0 < 256; k0 += 32) {
    v16u a = lds_afrag(&sHt[(wave * 16) * 256 + k0], 256, lane);
#pragma unroll
    for (int t = 0; t < 4; ++t) {
      v16u bf = lds_bfrag(&sB[k0 * 64 + t * 16], 64, lane);
      acc[t] = wmma_bf16(a, bf, acc[t]);
    }
  }

  size_t obase = (size_t)bx * (Pp * Nn);
#pragma unroll
  for (int t = 0; t < 4; ++t)
#pragma unroll
    for (int r = 0; r < 8; ++r) {
      int p = wave * 16 + r + ((lane >> 4) << 3);
      int n = 16 * t + (lane & 15);
      lst[obase + p * 64 + n] = acc[t][r];
    }
}

// ---------------- 5. inter-chunk recurrence (sequential over 8 chunks) ------
__global__ void recurrence_k(const float* __restrict__ lst,
                             const float* __restrict__ csum,
                             float* __restrict__ stin)
{
  int b = blockIdx.x >> 6, h = blockIdx.x & 63;
  int tid = threadIdx.x;
  float carry[16];
#pragma unroll
  for (int i = 0; i < 16; ++i) carry[i] = 0.f;
  for (int c = 0; c < NCc; ++c) {
    int idx = (b * NCc + c) * NHh + h;
    float g = fexp(csum[idx]);
    size_t off = (size_t)idx * (Pp * Nn);
#pragma unroll
    for (int i = 0; i < 16; ++i) {
      int e = i * 256 + tid;
      stin[off + e] = carry[i];
      carry[i] = carry[i] * g + lst[off + e];
    }
  }
}

// ---------------- 6. fused intra-chunk SSD: Y = (L*CB^T)hdt + C st^T e^A + Dh
__global__ __launch_bounds__(256) void ssd_intra_k(
    const float* __restrict__ conv, const float* __restrict__ dt,
    const float* __restrict__ acs, const float* __restrict__ stin,
    const float* __restrict__ Dp, float* __restrict__ y)
{
  extern __shared__ __align__(16) char smem[];
  ushort_t* sBt = (ushort_t*)smem;            // [n=64][s=256]   raw B^T
  ushort_t* sSt = sBt + 64 * 256;             // [n=64][p=64]    states_in^T
  ushort_t* sM  = sSt + 64 * 64;              // [8 waves][32][32]
  float*    sAcs = (float*)(sM + 8 * 32 * 32);// [256]

  int bx = blockIdx.x;
  int h = bx & 63, c = (bx >> 6) & 7, b = bx >> 9;
  int tid = threadIdx.x, lane = tid & 31, wave = tid >> 5;
  int abase = (b * NCc + c) * NHh + h;
  size_t rowbase = (size_t)(b * Ss + c * CSs);

  for (int i = 0; i < 64; ++i) {                       // B^T
    int idx = i * 256 + tid;
    int s = idx >> 6, n = idx & 63;
    sBt[n * 256 + s] = f2bf(conv[(rowbase + s) * CONVD + Ee + n]);
  }
  {
    size_t sb = (size_t)abase * (Pp * Nn);
    for (int i = 0; i < 16; ++i) {                     // states^T
      int idx = i * 256 + tid;
      int p = idx >> 6, n = idx & 63;
      sSt[n * 64 + p] = f2bf(stin[sb + idx]);
    }
  }
  sAcs[tid] = acs[(size_t)abase * CSs + tid];
  __syncthreads();

  const int l0 = wave * 32;

  // cache C A-fragments (rows l, K = n) straight from global fp32
  v16u aC[2][2];
#pragma unroll
  for (int i = 0; i < 2; ++i)
#pragma unroll
    for (int kk = 0; kk < 2; ++kk)
      aC[i][kk] = ga_afrag(conv + (rowbase + l0 + 16 * i) * CONVD + (Ee + Nn) + kk * 32,
                           CONVD, lane);

  // Y_off = C @ states^T
  v8f accY[2][4];
#pragma unroll
  for (int i = 0; i < 2; ++i)
#pragma unroll
    for (int t = 0; t < 4; ++t) accY[i][t] = vzero8();
#pragma unroll
  for (int kk = 0; kk < 2; ++kk)
#pragma unroll
    for (int t = 0; t < 4; ++t) {
      v16u bf = lds_bfrag(&sSt[kk * 32 * 64 + t * 16], 64, lane);
      accY[0][t] = wmma_bf16(aC[0][kk], bf, accY[0][t]);
      accY[1][t] = wmma_bf16(aC[1][kk], bf, accY[1][t]);
    }
  // scale by exp(A_cs[l])
#pragma unroll
  for (int i = 0; i < 2; ++i)
#pragma unroll
    for (int r = 0; r < 8; ++r) {
      int l = l0 + 16 * i + r + ((lane >> 4) << 3);
      float e = fexp(sAcs[l]);
#pragma unroll
      for (int t = 0; t < 4; ++t) accY[i][t][r] *= e;
    }

  // stream s-blocks of 32: Gm -> mask -> M(bf16, per-wave LDS) -> M @ hdt
  ushort_t* sMw = sM + wave * (32 * 32);
  for (int sb = 0; sb < 8; ++sb) {
    const int s0 = sb * 32;
    v8f accG[2][2];
#pragma unroll
    for (int i = 0; i < 2; ++i)
#pragma unroll
      for (int j = 0; j < 2; ++j) accG[i][j] = vzero8();
#pragma unroll
    for (int kk = 0; kk < 2; ++kk)
#pragma unroll
      for (int j = 0; j < 2; ++j) {
        v16u bf = lds_bfrag(&sBt[kk * 32 * 256 + s0 + 16 * j], 256, lane);
        accG[0][j] = wmma_bf16(aC[0][kk], bf, accG[0][j]);
        accG[1][j] = wmma_bf16(aC[1][kk], bf, accG[1][j]);
      }
    // causal decay mask, write bf16 M tile (wave-private LDS region)
#pragma unroll
    for (int i = 0; i < 2; ++i)
#pragma unroll
      for (int j = 0; j < 2; ++j)
#pragma unroll
        for (int r = 0; r < 8; ++r) {
          int ll = 16 * i + r + ((lane >> 4) << 3);
          int sl = 16 * j + (lane & 15);
          int l = l0 + ll, s = s0 + sl;
          float m = (s <= l) ? fexp(sAcs[l] - sAcs[s]) : 0.f;
          sMw[ll * 32 + sl] = f2bf(accG[i][j][r] * m);
        }
    asm volatile("s_wait_dscnt 0" ::: "memory");  // same-wave LDS RAW

    // hdt B-fragments from global (lane -> p column, elements -> s), scaled by dt
    const float* dcol = dt + (rowbase + s0) * NHh + h;
    const float* hrow = conv + (rowbase + s0) * CONVD + h * Pp;
    v16u bH[4];
#pragma unroll
    for (int t = 0; t < 4; ++t) bH[t] = gb_scaled(hrow + t * 16, CONVD, dcol, NHh, lane);

#pragma unroll
    for (int i = 0; i < 2; ++i) {
      v16u aM = lds_afrag(&sMw[16 * i * 32], 32, lane);
#pragma unroll
      for (int t = 0; t < 4; ++t) accY[i][t] = wmma_bf16(aM, bH[t], accY[i][t]);
    }
  }

  // epilogue: + D*h (pre-dt), store y
  float Dh = Dp[h];
#pragma unroll
  for (int i = 0; i < 2; ++i)
#pragma unroll
    for (int t = 0; t < 4; ++t)
#pragma unroll
      for (int r = 0; r < 8; ++r) {
        int l = l0 + 16 * i + r + ((lane >> 4) << 3);
        int p = 16 * t + (lane & 15);
        size_t row = rowbase + l;
        float v = accY[i][t][r] + Dh * conv[row * CONVD + h * Pp + p];
        y[row * Ee + h * Pp + p] = v;
      }
}

// ---------------- 7. gated RMSNorm (in-place on y) --------------------------
__global__ __launch_bounds__(256) void gate_norm_k(
    float* __restrict__ y, const float* __restrict__ proj,
    const float* __restrict__ nw)
{
  __shared__ float red[256];
  int row = blockIdx.x, tid = threadIdx.x;
  size_t yb = (size_t)row * Ee;
  size_t pb = (size_t)row * PROJ_;
  float s = 0.f;
  for (int j = tid; j < Ee; j += 256) {
    float yf = y[yb + j] * silu(proj[pb + j]);
    s += yf * yf;
  }
  red[tid] = s;
  __syncthreads();
  for (int o = 128; o > 0; o >>= 1) {
    if (tid < o) red[tid] += red[tid + o];
    __syncthreads();
  }
  float rs = rsqrtf(red[0] / (float)Ee + 1e-6f);
  for (int j = tid; j < Ee; j += 256) {
    float yf = y[yb + j] * silu(proj[pb + j]);
    y[yb + j] = nw[j] * yf * rs;
  }
}

// ---------------- launch ----------------------------------------------------
extern "C" void kernel_launch(void* const* d_in, const int* in_sizes, int n_in,
                              void* d_out, int out_size, void* d_ws, size_t ws_size,
                              hipStream_t stream)
{
  const float* x    = (const float*)d_in[0];
  const float* w1   = (const float*)d_in[1];
  const float* cw   = (const float*)d_in[2];
  const float* cb   = (const float*)d_in[3];
  const float* dtb  = (const float*)d_in[4];
  const float* alog = (const float*)d_in[5];
  const float* Dp   = (const float*)d_in[6];
  const float* nw   = (const float*)d_in[7];
  const float* w2   = (const float*)d_in[8];
  float* out = (float*)d_out;

  float* ws   = (float*)d_ws;
  float* proj = ws;                                          // 4096 x 8384
  float* conv = proj + (size_t)MROWS * PROJ_;                // 4096 x 4224
  float* dtv  = conv + (size_t)MROWS * CONVD;                // 4096 x 64
  float* acsv = dtv + (size_t)MROWS * NHh;                   // B*NC*NH*CS
  float* csum = acsv + (size_t)Bb * NCc * NHh * CSs;         // B*NC*NH
  float* lst  = csum + (size_t)Bb * NCc * NHh;               // B*NC*NH*P*N
  float* stin = lst + (size_t)Bb * NCc * NHh * Pp * Nn;      // B*NC*NH*P*N
  float* ybuf = stin + (size_t)Bb * NCc * NHh * Pp * Nn;     // 4096 x 4096

  // 1) proj = x @ in_proj_w^T
  gemm_nt_bf16<<<dim3((PROJ_ + 127) / 128, MROWS / 128), 256, 0, stream>>>(
      x, w1, proj, MROWS, PROJ_, Hh);

  // 2) conv + silu
  conv_silu_k<<<(int)(((size_t)Bb * Ss * CONVD + 255) / 256), 256, 0, stream>>>(
      proj, cw, cb, conv);

  // 3) dt softplus + A*dt cumsums
  dt_scan_k<<<(Bb * NCc * NHh + 255) / 256, 256, 0, stream>>>(
      proj, dtb, alog, dtv, acsv, csum);

  // 4) per-chunk local states (WMMA)
  chunk_states_k<<<Bb * NCc * NHh, 128, (64 * 256 + 256 * 64) * sizeof(ushort_t),
                   stream>>>(conv, dtv, acsv, csum, lst);

  // 5) inter-chunk recurrence
  recurrence_k<<<Bb * NHh, 256, 0, stream>>>(lst, csum, stin);

  // 6) fused intra-chunk SSD (WMMA)
  size_t smem2 = (size_t)(64 * 256 + 64 * 64 + 8 * 32 * 32) * sizeof(ushort_t)
               + 256 * sizeof(float);
  ssd_intra_k<<<Bb * NCc * NHh, 256, smem2, stream>>>(
      conv, dtv, acsv, stin, Dp, ybuf);

  // 7) gated RMSNorm (in place)
  gate_norm_k<<<MROWS, 256, 0, stream>>>(ybuf, proj, nw);

  // 8) out = yn @ out_proj_w^T
  gemm_nt_bf16<<<dim3(Hh / 128, MROWS / 128), 256, 0, stream>>>(
      ybuf, w2, out, MROWS, Hh, Ee);
}
